// HierarchicalGNNBranch_50964081935228
// MI455X (gfx1250) — matrix operand order repaired and checked by reading, not compiled, not run on hardware
//
#include <hip/hip_runtime.h>
#include <stdint.h>

#define N_NODES 20000
#define IN_DIM  3072
#define N_GRAPHS 128
#define BN_EPS  1e-5f

// ---------------------------------------------------------------------------
// Types for CDNA5 WMMA (gfx1250, wave32)
// ---------------------------------------------------------------------------
typedef __attribute__((ext_vector_type(16))) __bf16 v16bf;
typedef __attribute__((ext_vector_type(8)))  float  v8f;

union FragBF { v16bf v; uint4 u[2]; };

__device__ __forceinline__ unsigned short f2bf(float f) {
  unsigned int u = __float_as_uint(f);
  u += 0x7FFFu + ((u >> 16) & 1u);        // round-to-nearest-even
  return (unsigned short)(u >> 16);
}

// ---------------------------------------------------------------------------
// Elementwise helpers
// ---------------------------------------------------------------------------
__global__ void fill_f32(float* __restrict__ p, float v, long n) {
  long i = (long)blockIdx.x * blockDim.x + threadIdx.x;
  if (i < n) p[i] = v;
}

__global__ void cvt_f32_bf16(const float* __restrict__ x,
                             unsigned short* __restrict__ y, long n) {
  long i = (long)blockIdx.x * blockDim.x + threadIdx.x;
  if (i < n) y[i] = f2bf(x[i]);
}

// Store W (din x dout, row-major fp32) as Bt (rows x cols, row-major bf16):
// Bt[n,k] = W[k,n], zero padded for n >= dout or k >= din.
__global__ void weight_transpose_bf16(const float* __restrict__ w, int din, int dout,
                                      unsigned short* __restrict__ bt, int rows, int cols) {
  long i = (long)blockIdx.x * blockDim.x + threadIdx.x;
  long total = (long)rows * cols;
  if (i >= total) return;
  int n = (int)(i / cols);
  int k = (int)(i - (long)n * cols);
  float v = (n < dout && k < din) ? w[(long)k * dout + n] : 0.0f;
  bt[i] = f2bf(v);
}

// ---------------------------------------------------------------------------
// WMMA GEMM: C[M x NP] = A[M x KP] * Bt[NP x KP]^T
//   A row-major bf16 (lda = KP elems), Bt N-major/K-contig bf16 (ldb = KP elems)
//   Each wave computes a 32x64 tile: 2x4 grid of 16x16 WMMA accumulators
//   (8 WMMAs per K=32 step from 12 b128 loads -> 1.5 loads/WMMA; halves the
//   L2 re-read traffic of A compared to a 32x32 tile).
//   EPI==0: out bf16 = relu(C + bias)   EPI==1: out fp32 = C (no bias)
// Requires: M % 32 == 0, NP % 64 == 0, KP % 32 == 0.
// ---------------------------------------------------------------------------
template <int EPI>
__global__ void gemm_wmma_bf16(const unsigned short* __restrict__ A, int lda,
                               const unsigned short* __restrict__ Bt, int ldb,
                               void* __restrict__ C, int ldc,
                               const float* __restrict__ bias, int nbias,
                               int Mt, int Nt, int KP) {
  int wv = (int)(((long)blockIdx.x * blockDim.x + threadIdx.x) >> 5);
  if (wv >= Mt * Nt) return;                 // wave-uniform: EXEC stays all-1s
  int lane = threadIdx.x & 31;
  int mt = wv / Nt;
  int nt = wv - mt * Nt;
  int half = lane >> 4;                      // 0: lanes 0-15, 1: lanes 16-31
  int lr   = lane & 15;

  // A fragment rows (16-bit A 16x32 layout, ISA 7.12.2):
  //   lane<16 : elems 0..7 = K+0..7,  elems 8..15 = K+16..23
  //   lane>=16: elems 0..7 = K+8..15, elems 8..15 = K+24..31
  const unsigned short* a0p = A + (long)(mt * 32 + lr) * lda;
  // B fragment (32x16 16-bit): same K order as A: lanes 0-15 elems 0..7 = K+0..7,
  //   8..15 = K+16..23; lanes 16-31 elems 0..7 = K+8..15, 8..15 = K+24..31.
  const unsigned short* b0p = Bt + (long)(nt * 64 + lr) * ldb;
  const int ka = half * 8;
  const int kb = half * 8;

  v8f acc[2][4] = {};
  for (int k0 = 0; k0 < KP; k0 += 32) {
    FragBF a[2], b[4];
#pragma unroll
    for (int i = 0; i < 2; ++i) {
      const unsigned short* ap = a0p + (long)i * 16 * lda + k0 + ka;
      a[i].u[0] = *(const uint4*)(ap);
      a[i].u[1] = *(const uint4*)(ap + 16);
    }
#pragma unroll
    for (int j = 0; j < 4; ++j) {
      const unsigned short* bp = b0p + (long)j * 16 * ldb + k0 + kb;
      b[j].u[0] = *(const uint4*)(bp);
      b[j].u[1] = *(const uint4*)(bp + 16);
    }
    __builtin_prefetch(a0p + k0 + 64, 0, 3);   // global_prefetch_b8
    __builtin_prefetch(b0p + k0 + 64, 0, 3);
#pragma unroll
    for (int i = 0; i < 2; ++i)
#pragma unroll
      for (int j = 0; j < 4; ++j)
        acc[i][j] = __builtin_amdgcn_wmma_f32_16x16x32_bf16(
            false, a[i].v, false, b[j].v, (short)0, acc[i][j], false, false);
  }

  // C/D layout: VGPR r holds M = r + 8*half, N = lane % 16.
#pragma unroll
  for (int i = 0; i < 2; ++i) {
#pragma unroll
    for (int j = 0; j < 4; ++j) {
      int nn = nt * 64 + j * 16 + lr;
      float bv = 0.0f;
      if (EPI == 0) bv = (nn < nbias) ? bias[nn] : 0.0f;
      int mbase = mt * 32 + i * 16 + half * 8;
#pragma unroll
      for (int r = 0; r < 8; ++r) {
        float v = acc[i][j][r];
        if (EPI == 0) {
          v = fmaxf(v + bv, 0.0f);
          ((unsigned short*)C)[(long)(mbase + r) * ldc + nn] = f2bf(v);
        } else {
          ((float*)C)[(long)(mbase + r) * ldc + nn] = v;
        }
      }
    }
  }
}

// ---------------------------------------------------------------------------
// Graph kernels
// ---------------------------------------------------------------------------
__global__ void deg_count(const int* __restrict__ dst, float* __restrict__ deg, int E) {
  int e = blockIdx.x * blockDim.x + threadIdx.x;
  if (e < E) atomicAdd(&deg[dst[e]], 1.0f);
}

__global__ void deg_to_dinv(float* __restrict__ d, int n) {
  int i = blockIdx.x * blockDim.x + threadIdx.x;
  if (i < n) d[i] = rsqrtf(d[i] + 1.0f);     // +1 self loop; always > 0
}

// One wave per edge, lanes stride the feature dim; f32 atomics into AGG.
__global__ void edge_agg(const int* __restrict__ src, const int* __restrict__ dst,
                         const float* __restrict__ dinv, const float* __restrict__ h,
                         float* __restrict__ agg, int E, int fd, int ld) {
  int wv = (int)(((long)blockIdx.x * blockDim.x + threadIdx.x) >> 5);
  if (wv >= E) return;
  int lane = threadIdx.x & 31;
  int s = src[wv], d = dst[wv];
  float w = dinv[s] * dinv[d];
  const float* hs = h + (long)s * ld;
  float* ad = agg + (long)d * ld;
  for (int f = lane; f < fd; f += 32) atomicAdd(&ad[f], hs[f] * w);
}

// Self-loop + GCN bias + BatchNorm(eval) + ReLU; emit bf16 next-layer input
// (zero in pad columns) and optionally fp32 (in-place into AGG) for pooling.
__global__ void bn_relu(const float* __restrict__ agg, const float* __restrict__ h3,
                        const float* __restrict__ dinv, const float* __restrict__ gb,
                        const float* __restrict__ gamma, const float* __restrict__ beta,
                        const float* __restrict__ mean, const float* __restrict__ var,
                        unsigned short* __restrict__ xout, float* __restrict__ ffinal,
                        int fd, int ld, int writeFinal) {
  long i = (long)blockIdx.x * blockDim.x + threadIdx.x;
  long total = (long)N_NODES * ld;
  if (i >= total) return;
  int node = (int)(i / ld);
  int f    = (int)(i - (long)node * ld);
  float v = 0.0f;
  if (f < fd) {
    float di = dinv[node];
    v = agg[i] + h3[i] * di * di + gb[f];
    v = (v - mean[f]) * rsqrtf(var[f] + BN_EPS) * gamma[f] + beta[f];
    v = fmaxf(v, 0.0f);
  }
  xout[i] = f2bf(v);
  if (writeFinal) ffinal[i] = v;
}

// global_max_pool: post-ReLU values >= 0, so uint atomicMax over zero-init out
// matches float max ordering.
__global__ void seg_max(const float* __restrict__ xf, const int* __restrict__ batch,
                        float* __restrict__ out, int fd, int ld) {
  long i = (long)blockIdx.x * blockDim.x + threadIdx.x;
  long total = (long)N_NODES * fd;
  if (i >= total) return;
  int node = (int)(i / fd);
  int f    = (int)(i - (long)node * fd);
  float v = xf[(long)node * ld + f];
  atomicMax((unsigned int*)&out[(long)batch[node] * fd + f], __float_as_uint(v));
}

// ---------------------------------------------------------------------------
// Host orchestration
// ---------------------------------------------------------------------------
static inline dim3 g1(long n) { return dim3((unsigned)((n + 255) / 256)); }

extern "C" void kernel_launch(void* const* d_in, const int* in_sizes, int n_in,
                              void* d_out, int out_size, void* d_ws, size_t ws_size,
                              hipStream_t stream) {
  (void)n_in; (void)ws_size;
  const float* x    = (const float*)d_in[0];
  const int*   ei   = (const int*)d_in[1];
  const int*   bat  = (const int*)d_in[2];
  const int    E    = in_sizes[1] / 2;
  const int*   srcI = ei;
  const int*   dstI = ei + E;

  static const int FD[4]   = {80, 160, 400, 600};
  static const int FD4[4]  = {320, 640, 1600, 2400};   // true hidden dims
  static const int FD4P[4] = {320, 640, 1600, 2432};   // ceil64(4*fd)
  static const int NPF[4]  = {128, 192, 448, 640};     // ceil64(fd)

  // workspace layout (256B aligned)
  char*  ws  = (char*)d_ws;
  size_t off = 0;
  auto take = [&](size_t bytes) -> char* {
    char* p = ws + off;
    off += (bytes + 255) & ~(size_t)255;
    return p;
  };
  unsigned short* XBF = (unsigned short*)take((size_t)N_NODES * IN_DIM * 2); // activations (bf16)
  unsigned short* H1  = (unsigned short*)take((size_t)N_NODES * 2432 * 2);
  unsigned short* H2  = (unsigned short*)take((size_t)N_NODES * 640 * 2);
  float*          H3  = (float*)take((size_t)N_NODES * 640 * 4);             // pre-aggregation
  float*          AGG = (float*)take((size_t)N_NODES * 640 * 4);             // scatter target / final fp32
  unsigned short* BT  = (unsigned short*)take((size_t)8 << 20);              // transposed bf16 weights
  float*          DINV = (float*)take((size_t)N_NODES * 4);

  const int Mt = N_NODES / 32;   // 625, exact

  // Degrees -> D^-1/2 (self loop included as +1)
  fill_f32<<<g1(N_NODES), 256, 0, stream>>>(DINV, 0.0f, N_NODES);
  deg_count<<<g1(E), 256, 0, stream>>>(dstI, DINV, E);
  deg_to_dinv<<<g1(N_NODES), 256, 0, stream>>>(DINV, N_NODES);

  // x -> bf16 once (halves layer-0 A traffic: 123 MB instead of 246 MB)
  cvt_f32_bf16<<<g1((long)N_NODES * IN_DIM), 256, 0, stream>>>(x, XBF, (long)N_NODES * IN_DIM);

  for (int L = 0; L < 4; ++L) {
    const float* wa    = (const float*)d_in[3 + 10 * L + 0];
    const float* ba    = (const float*)d_in[3 + 10 * L + 1];
    const float* wb    = (const float*)d_in[3 + 10 * L + 2];
    const float* bb    = (const float*)d_in[3 + 10 * L + 3];
    const float* gw    = (const float*)d_in[3 + 10 * L + 4];
    const float* gb    = (const float*)d_in[3 + 10 * L + 5];
    const float* gamma = (const float*)d_in[3 + 10 * L + 6];
    const float* beta  = (const float*)d_in[3 + 10 * L + 7];
    const float* mean  = (const float*)d_in[3 + 10 * L + 8];
    const float* var   = (const float*)d_in[3 + 10 * L + 9];

    const int fd   = FD[L];
    const int fd4  = FD4[L];
    const int fd4p = FD4P[L];
    const int npf  = NPF[L];
    const int dinr = (L == 0) ? IN_DIM : FD[L - 1];    // true K
    const int dinp = (L == 0) ? IN_DIM : NPF[L - 1];   // padded K (mult of 32)

    // GEMM 1: h1 = relu(x @ wa + ba)   [N x dinp] * [dinp x fd4p]
    weight_transpose_bf16<<<g1((long)fd4p * dinp), 256, 0, stream>>>(wa, dinr, fd4, BT, fd4p, dinp);
    {
      long waves = (long)Mt * (fd4p / 64);
      gemm_wmma_bf16<0><<<dim3((unsigned)((waves + 7) / 8)), 256, 0, stream>>>(
          XBF, dinp, BT, dinp, H1, fd4p, ba, fd4, Mt, fd4p / 64, dinp);
    }

    // GEMM 2: h2 = relu(h1 @ wb + bb)  [N x fd4p] * [fd4p x npf]; pad cols -> 0
    weight_transpose_bf16<<<g1((long)npf * fd4p), 256, 0, stream>>>(wb, fd4, fd, BT, npf, fd4p);
    {
      long waves = (long)Mt * (npf / 64);
      gemm_wmma_bf16<0><<<dim3((unsigned)((waves + 7) / 8)), 256, 0, stream>>>(
          H1, fd4p, BT, fd4p, H2, npf, bb, fd, Mt, npf / 64, fd4p);
    }

    // GEMM 3: h3 = h2 @ gw (fp32 out, no bias)  [N x npf] * [npf x npf]
    weight_transpose_bf16<<<g1((long)npf * npf), 256, 0, stream>>>(gw, fd, fd, BT, npf, npf);
    {
      long waves = (long)Mt * (npf / 64);
      gemm_wmma_bf16<1><<<dim3((unsigned)((waves + 7) / 8)), 256, 0, stream>>>(
          H2, npf, BT, npf, H3, npf, nullptr, 0, Mt, npf / 64, npf);
    }

    // Message passing: agg = scatter_add(h3[src] * norm, dst); self loop in BN pass
    fill_f32<<<g1((long)N_NODES * npf), 256, 0, stream>>>(AGG, 0.0f, (long)N_NODES * npf);
    edge_agg<<<dim3((unsigned)(((long)E + 7) / 8)), 256, 0, stream>>>(
        srcI, dstI, DINV, H3, AGG, E, fd, npf);

    // + gb, BatchNorm(eval), ReLU; next-layer bf16 input (zero-padded cols)
    bn_relu<<<g1((long)N_NODES * npf), 256, 0, stream>>>(
        AGG, H3, DINV, gb, gamma, beta, mean, var, XBF, AGG, fd, npf, (L == 3) ? 1 : 0);
  }

  // global_max_pool over sorted batch -> (G, 600)
  fill_f32<<<g1((long)out_size), 256, 0, stream>>>((float*)d_out, 0.0f, (long)out_size);
  seg_max<<<g1((long)N_NODES * FD[3]), 256, 0, stream>>>(
      AGG, bat, (float*)d_out, FD[3], NPF[3]);
}
